// FasterRCNN_87943750352918
// MI455X (gfx1250) — matrix operand order repaired and checked by reading, not compiled, not run on hardware
//
#include <hip/hip_runtime.h>

// ============================================================================
// Faster-RCNN head for MI455X (gfx1250, wave32, WMMA).
//
// All GEMM-shaped stages (patchify conv, 3x3 rpn conv, 1x1 heads, FC) use
// v_wmma_f32_16x16x32_f16: fp16 A/B fragments, fp32 accumulation.
//
// FC (M=40 pad 48, K=50176, N=1024) is the only HBM-bound stage (fc_w=205MB,
// ~20 FLOP/B): the block cooperatively stages each 32Kx128N fc_w chunk into
// LDS with coalesced float4 loads (one clean pass over fc_w), stages the
// 48x32 fp16 A chunk, and 8 waves build WMMA fragments from LDS.
//
// Output layout (floats, concatenated):
//   rpn_reg  [2,9216,4]  @ 0
//   rpn_cls  [2,9216]    @ 73728
//   nms_reg  [2,20,4]    @ 92160
//   nms_cls  [2,20]      @ 92320
//   rcnn_reg [2,20,4]    @ 92360
//   rcnn_cls [2,20,21]   @ 92520
//   anchors  [9216,4]    @ 93360   (total 130224)
//
// Workspace (~17 MB): fm f32 | h f32 | proposals f32 | pooled f16 | hf f32
// ============================================================================

typedef __attribute__((ext_vector_type(16))) _Float16 v16h;
typedef __attribute__((ext_vector_type(8)))  float    v8f;

#define WMMA_F16(A, B, C) \
  __builtin_amdgcn_wmma_f32_16x16x32_f16(false, (A), false, (B), (short)0, (C), false, false)

// ---- output offsets (floats) ----
#define OFF_RPN_REG   0
#define OFF_RPN_CLS   73728
#define OFF_NMS_REG   92160
#define OFF_NMS_CLS   92320
#define OFF_RCNN_REG  92360
#define OFF_RCNN_CLS  92520
#define OFF_ANCHORS   93360

// ---- workspace offsets (bytes) ----
#define WS_FM      0u           // f32 [2*1024*32*32]  = 8388608 B
#define WS_H       8388608u     // f32 [2*512*32*32]   = 4194304 B
#define WS_PROP    12582912u    // f32 [2*9216*4]      = 294912 B
#define WS_POOLED  12877824u    // f16 [40*50176]      = 4014080 B
#define WS_HF      16891904u    // f32 [40*1024]       = 163840 B

#define NEGV (-1e30f)

// A-fragment K index for element e, lane-half hf (CDNA5 16-bit A 16x32 layout)
__device__ __forceinline__ int a_klocal(int e, int hf) {
  return ((e < 8) ? 0 : 16) + hf * 8 + (e & 7);
}
// B-fragment K index (same order as A: lanes 0-15: K=0..7,16..23; lanes 16-31: K=8..15,24..31)
__device__ __forceinline__ int b_klocal(int e, int hf) {
  return ((e < 8) ? 0 : 16) + hf * 8 + (e & 7);
}

__device__ __forceinline__ float anchor_coord(int y, int x, int a, int coord) {
  const int si = a / 3, ri = a - si * 3;
  const float s  = (float)(64 << si);
  const float sr = (ri == 0) ? 0.70710678118654752f : (ri == 1 ? 1.0f : 1.41421356237309505f);
  const float w = s * sr, h = s / sr;
  const float cx = (x + 0.5f) * 16.0f, cy = (y + 0.5f) * 16.0f;
  if (coord == 0) return cx - 0.5f * w;
  if (coord == 1) return cy - 0.5f * h;
  if (coord == 2) return cx + 0.5f * w;
  return cy + 0.5f * h;
}

// ---------------------------------------------------------------------------
// anchors -> d_out
__global__ __launch_bounds__(256) void k_anchors(float* __restrict__ out) {
  int i = blockIdx.x * 256 + threadIdx.x;
  if (i >= 9216 * 4) return;
  int ai = i >> 2, coord = i & 3;
  int cell = ai / 9, a = ai - cell * 9;
  int y = cell >> 5, x = cell & 31;
  out[i] = anchor_coord(y, x, a, coord);
}

// ---------------------------------------------------------------------------
// K1: stride-16 16x16 VALID conv + relu  ==  GEMM M=2048 (b,y,x) x K=768 x N=1024
// one wave per 16x16 C tile; 24 K-chunks of 32 -> 24 wmma per wave.
__global__ __launch_bounds__(256) void k_patch_conv(const float* __restrict__ img,
                                                    const float* __restrict__ w,
                                                    const float* __restrict__ bias,
                                                    float* __restrict__ fm) {
  const int lane = threadIdx.x & 31, wv = threadIdx.x >> 5;
  const int gw = blockIdx.x * 8 + wv;      // 8192 waves
  const int mT = gw >> 6, nT = gw & 63;    // 128 x 64 tiles
  const int hf = lane >> 4, l15 = lane & 15;
  const int mBase = mT * 16, nCol = nT * 16 + l15;

  // per-lane A row
  const int m = mBase + l15;
  const int b = m >> 10, rem = m & 1023, y = rem >> 5, x = rem & 31;
  const float* imgB = img + (size_t)b * 3 * 512 * 512;

  v8f acc = {};
#pragma unroll 1
  for (int kc = 0; kc < 768; kc += 32) {
    v16h afr, bfr;
#pragma unroll
    for (int e = 0; e < 16; ++e) {
      int ka = kc + a_klocal(e, hf);
      int ci = ka >> 8, r = ka & 255, ky = r >> 4, kx = r & 15;
      afr[e] = (_Float16)imgB[((size_t)ci * 512 + (y * 16 + ky)) * 512 + (x * 16 + kx)];
      int kb = kc + b_klocal(e, hf);
      bfr[e] = (_Float16)w[(size_t)nCol * 768 + kb];
    }
    acc = WMMA_F16(afr, bfr, acc);
  }
  const float bv = bias[nCol];
#pragma unroll
  for (int v = 0; v < 8; ++v) {
    int row = hf * 8 + v;
    int mm = mBase + row;
    int bb = mm >> 10, rr = mm & 1023, yy = rr >> 5, xx = rr & 31;
    float val = acc[v] + bv;
    fm[(((size_t)bb * 1024 + nCol) * 32 + yy) * 32 + xx] = val > 0.f ? val : 0.f;
  }
}

// ---------------------------------------------------------------------------
// K2: 3x3 SAME conv + relu == GEMM M=2048 x K=9216 x N=512 (im2col gather)
__global__ __launch_bounds__(256) void k_rpn_conv(const float* __restrict__ fm,
                                                  const float* __restrict__ w,
                                                  const float* __restrict__ bias,
                                                  float* __restrict__ h) {
  const int lane = threadIdx.x & 31, wv = threadIdx.x >> 5;
  const int gw = blockIdx.x * 8 + wv;      // 4096 waves
  const int mT = gw >> 5, nT = gw & 31;    // 128 x 32 tiles
  const int hf = lane >> 4, l15 = lane & 15;
  const int mBase = mT * 16, nCol = nT * 16 + l15;

  const int m = mBase + l15;
  const int b = m >> 10, rem = m & 1023, y = rem >> 5, x = rem & 31;
  const float* fmB = fm + (size_t)b * 1024 * 32 * 32;

  v8f acc = {};
#pragma unroll 1
  for (int kc = 0; kc < 9216; kc += 32) {
    v16h afr, bfr;
#pragma unroll
    for (int e = 0; e < 16; ++e) {
      int ka = kc + a_klocal(e, hf);
      int ci = ka / 9, r = ka - ci * 9;
      int ky = r / 3, kx = r - ky * 3;
      int yy = y + ky - 1, xx = x + kx - 1;
      bool ok = (yy >= 0) & (yy < 32) & (xx >= 0) & (xx < 32);
      afr[e] = ok ? (_Float16)fmB[((size_t)ci * 32 + yy) * 32 + xx] : (_Float16)0;
      int kb = kc + b_klocal(e, hf);
      bfr[e] = (_Float16)w[(size_t)nCol * 9216 + kb];
    }
    __builtin_prefetch(&w[(size_t)nCol * 9216 + kc + 32], 0, 0);
    acc = WMMA_F16(afr, bfr, acc);
  }
  const float bv = bias[nCol];
#pragma unroll
  for (int v = 0; v < 8; ++v) {
    int row = hf * 8 + v;
    int mm = mBase + row;
    int bb = mm >> 10, rr = mm & 1023, yy = rr >> 5, xx = rr & 31;
    float val = acc[v] + bv;
    h[(((size_t)bb * 512 + nCol) * 32 + yy) * 32 + xx] = val > 0.f ? val : 0.f;
  }
}

// ---------------------------------------------------------------------------
// K3: 1x1 head convs == GEMM M=2048 x K=512 x N=45 (padded to 48).
// Writes rpn_reg / rpn_cls outputs and proposals (= rpn_reg + anchors).
__global__ __launch_bounds__(256) void k_heads(const float* __restrict__ h,
                                               const float* __restrict__ reg_w,
                                               const float* __restrict__ reg_b,
                                               const float* __restrict__ cls_w,
                                               const float* __restrict__ cls_b,
                                               float* __restrict__ o_reg,
                                               float* __restrict__ o_cls,
                                               float* __restrict__ prop) {
  const int lane = threadIdx.x & 31, wv = threadIdx.x >> 5;
  const int gw = blockIdx.x * 8 + wv;      // 384 waves
  const int mT = gw / 3, nT = gw - mT * 3; // 128 x 3 tiles
  const int hf = lane >> 4, l15 = lane & 15;
  const int mBase = mT * 16, nCol = nT * 16 + l15;

  const int m = mBase + l15;
  const int b = m >> 10, rem = m & 1023, y = rem >> 5, x = rem & 31;
  const float* hB = h + (size_t)b * 512 * 32 * 32;

  v8f acc = {};
#pragma unroll 1
  for (int kc = 0; kc < 512; kc += 32) {
    v16h afr, bfr;
#pragma unroll
    for (int e = 0; e < 16; ++e) {
      int ka = kc + a_klocal(e, hf);
      afr[e] = (_Float16)hB[((size_t)ka * 32 + y) * 32 + x];
      int kb = kc + b_klocal(e, hf);
      float wv2 = 0.f;
      if (nCol < 36)      wv2 = reg_w[(size_t)nCol * 512 + kb];
      else if (nCol < 45) wv2 = cls_w[(size_t)(nCol - 36) * 512 + kb];
      bfr[e] = (_Float16)wv2;
    }
    acc = WMMA_F16(afr, bfr, acc);
  }
  const float bv = (nCol < 36) ? reg_b[nCol] : (nCol < 45 ? cls_b[nCol - 36] : 0.f);
#pragma unroll
  for (int v = 0; v < 8; ++v) {
    int row = hf * 8 + v;
    int mm = mBase + row;
    int bb = mm >> 10, rr = mm & 1023, yy = rr >> 5, xx = rr & 31;
    int cell = yy * 32 + xx;
    float val = acc[v] + bv;
    if (nCol < 36) {
      int a = nCol >> 2, coord = nCol & 3;
      size_t idx = ((size_t)bb * 9216 + (size_t)cell * 9 + a) * 4 + coord;
      o_reg[idx] = val;
      prop[idx]  = val + anchor_coord(yy, xx, a, coord);
    } else if (nCol < 45) {
      o_cls[(size_t)bb * 9216 + (size_t)cell * 9 + (nCol - 36)] = val;
    }
  }
}

// ---------------------------------------------------------------------------
// K4: NMS, one workgroup per image. Scores in LDS, 20 x (argmax-reduce + IoU).
__global__ __launch_bounds__(256) void k_nms(const float* __restrict__ prop,
                                             const float* __restrict__ scores,
                                             float* __restrict__ o_box,
                                             float* __restrict__ o_sc) {
  __shared__ float s_sc[9216];
  __shared__ float rv[256];
  __shared__ int   ri[256];
  __shared__ float selb[4];
  __shared__ int   seli;
  __shared__ float selv;

  const int b = blockIdx.x, tid = threadIdx.x;
  const float* pb = prop + (size_t)b * 9216 * 4;
  for (int i = tid; i < 9216; i += 256) s_sc[i] = scores[(size_t)b * 9216 + i];
  __syncthreads();

  for (int it = 0; it < 20; ++it) {
    float bestv = -__builtin_inff();
    int   besti = 0x7fffffff;
    for (int i = tid; i < 9216; i += 256) {
      float v = s_sc[i];
      if (v > bestv || (v == bestv && i < besti)) { bestv = v; besti = i; }
    }
    rv[tid] = bestv; ri[tid] = besti;
    __syncthreads();
    for (int off = 128; off > 0; off >>= 1) {
      if (tid < off) {
        if (rv[tid + off] > rv[tid] ||
            (rv[tid + off] == rv[tid] && ri[tid + off] < ri[tid])) {
          rv[tid] = rv[tid + off]; ri[tid] = ri[tid + off];
        }
      }
      __syncthreads();
    }
    if (tid == 0) {
      int i = ri[0];
      selb[0] = pb[(size_t)i * 4 + 0]; selb[1] = pb[(size_t)i * 4 + 1];
      selb[2] = pb[(size_t)i * 4 + 2]; selb[3] = pb[(size_t)i * 4 + 3];
      seli = i; selv = rv[0];
      float* ob = o_box + ((size_t)b * 20 + it) * 4;
      ob[0] = selb[0]; ob[1] = selb[1]; ob[2] = selb[2]; ob[3] = selb[3];
      o_sc[(size_t)b * 20 + it] = selv;
    }
    __syncthreads();
    const float b0 = selb[0], b1 = selb[1], b2 = selb[2], b3 = selb[3];
    const float ab = fmaxf(b2 - b0, 0.f) * fmaxf(b3 - b1, 0.f);
    const int si = seli;
    for (int i = tid; i < 9216; i += 256) {
      float c0 = pb[(size_t)i * 4 + 0], c1 = pb[(size_t)i * 4 + 1];
      float c2 = pb[(size_t)i * 4 + 2], c3 = pb[(size_t)i * 4 + 3];
      float ix1 = fmaxf(b0, c0), iy1 = fmaxf(b1, c1);
      float ix2 = fminf(b2, c2), iy2 = fminf(b3, c3);
      float inter = fmaxf(ix2 - ix1, 0.f) * fmaxf(iy2 - iy1, 0.f);
      float ar = fmaxf(c2 - c0, 0.f) * fmaxf(c3 - c1, 0.f);
      float iou = inter / (ab + ar - inter + 1e-8f);
      if (iou > 0.3f || i == si) s_sc[i] = NEGV;
    }
    __syncthreads();
  }
}

// ---------------------------------------------------------------------------
// K5: ROI-align. One block per (roi, py, px); threads sweep 1024 channels.
// Emits fp16 directly (A-matrix of the FC GEMM), layout [roi][c*49+pix].
__global__ __launch_bounds__(256) void k_roi(const float* __restrict__ fm,
                                             const float* __restrict__ boxes,
                                             const int* __restrict__ img_id,
                                             _Float16* __restrict__ pooled) {
  const int q = blockIdx.x;                // 40*49 blocks
  const int roi = q / 49, pix = q - roi * 49;
  const int py = pix / 7, px = pix - py * 7;
  const int b = roi / 20;
  const int bimg = img_id[b];

  const float* bp = boxes + (size_t)roi * 4;
  const float b0 = bp[0] / 16.f, b1 = bp[1] / 16.f, b2 = bp[2] / 16.f, b3 = bp[3] / 16.f;
  const float tx = (px + 0.5f) / 7.f, ty = (py + 0.5f) / 7.f;
  float xs = b0 + fmaxf(b2 - b0, 0.f) * tx;
  float ys = b1 + fmaxf(b3 - b1, 0.f) * ty;
  float xcl = fminf(fmaxf(xs, 0.f), 31.f);
  float ycl = fminf(fmaxf(ys, 0.f), 31.f);
  float x0f = floorf(xcl), y0f = floorf(ycl);
  int ix0 = (int)x0f, iy0 = (int)y0f;
  int ix1 = min(ix0 + 1, 31), iy1 = min(iy0 + 1, 31);
  float lx = xcl - x0f, ly = ycl - y0f;
  const float w00 = (1.f - ly) * (1.f - lx), w01 = (1.f - ly) * lx;
  const float w10 = ly * (1.f - lx),         w11 = ly * lx;

  const float* fb = fm + (size_t)bimg * 1024 * 32 * 32;
  for (int c = threadIdx.x; c < 1024; c += 256) {
    const float* p = fb + (size_t)c * 32 * 32;
    float v = p[iy0 * 32 + ix0] * w00 + p[iy0 * 32 + ix1] * w01 +
              p[iy1 * 32 + ix0] * w10 + p[iy1 * 32 + ix1] * w11;
    pooled[(size_t)roi * 50176 + (size_t)c * 49 + pix] = (_Float16)v;
  }
}

// ---------------------------------------------------------------------------
// K6: FC GEMM  hf = relu(pooled @ fc_w + fc_b). M=48(pad), K=50176, N=1024.
// HBM-bound on fc_w (205 MB, ~20 FLOP/B): the block stages each 32K x 128N
// fc_w chunk into LDS with coalesced float4 loads (fc_w streamed exactly
// once, full cachelines), plus the 48x32 f16 A chunk (cuts per-wave L2
// traffic 8x). 8 waves each build fragments from LDS and run 3 WMMAs
// (3 M-tiles share one B fragment).
__global__ __launch_bounds__(256) void k_fc(const _Float16* __restrict__ pooled,
                                            const float* __restrict__ fcw,
                                            const float* __restrict__ fcb,
                                            float* __restrict__ hfo) {
  __shared__ __align__(16) float    s_b[32 * 128];  // [k][n] 16 KB
  __shared__ __align__(16) _Float16 s_a[48 * 32];   // [m][k]  3 KB

  const int tid  = threadIdx.x;
  const int lane = tid & 31, wv = tid >> 5;
  const int nb0  = blockIdx.x * 128;       // block covers 128 N columns
  const int hf   = lane >> 4, l15 = lane & 15;
  const int nCol = nb0 + wv * 16 + l15;

  v8f acc0 = {}, acc1 = {}, acc2 = {};
#pragma unroll 1
  for (int kc = 0; kc < 50176; kc += 32) {
    // ---- stage B chunk: 32 rows x 128 cols, coalesced float4 ----
#pragma unroll
    for (int i = 0; i < 4; ++i) {
      int f = tid + i * 256;               // 1024 float4s
      int row = f >> 5, c4 = (f & 31) * 4;
      const float4 v = *(const float4*)(fcw + (size_t)(kc + row) * 1024 + nb0 + c4);
      *(float4*)(s_b + row * 128 + c4) = v;
    }
    // ---- stage A chunk: 48 rows x 32 k (rows >= 40 are zero pad) ----
#pragma unroll
    for (int i = 0; i < 6; ++i) {
      int f = tid + i * 256;               // 1536 halves
      int row = f >> 5, kk = f & 31;
      s_a[row * 32 + kk] =
          (row < 40) ? pooled[(size_t)row * 50176 + kc + kk] : (_Float16)0;
    }
    __builtin_prefetch(fcw + (size_t)(kc + 32 + (tid >> 5)) * 1024 + nb0 + (tid & 31) * 4, 0, 0);
    __syncthreads();

    // ---- build fragments from LDS, 3 WMMAs sharing one B fragment ----
    v16h bfr, a0, a1, a2;
#pragma unroll
    for (int e = 0; e < 16; ++e) {
      bfr[e] = (_Float16)s_b[b_klocal(e, hf) * 128 + wv * 16 + l15];
      int ka = a_klocal(e, hf);
      a0[e] = s_a[(l15)      * 32 + ka];
      a1[e] = s_a[(16 + l15) * 32 + ka];
      a2[e] = s_a[(32 + l15) * 32 + ka];
    }
    acc0 = WMMA_F16(a0, bfr, acc0);
    acc1 = WMMA_F16(a1, bfr, acc1);
    acc2 = WMMA_F16(a2, bfr, acc2);
    __syncthreads();
  }
  const float bv = fcb[nCol];
#pragma unroll
  for (int v = 0; v < 8; ++v) {
    int row = hf * 8 + v;
    float v0 = acc0[v] + bv, v1 = acc1[v] + bv, v2 = acc2[v] + bv;
    hfo[(size_t)row * 1024 + nCol]        = v0 > 0.f ? v0 : 0.f;
    hfo[(size_t)(16 + row) * 1024 + nCol] = v1 > 0.f ? v1 : 0.f;
    if (32 + row < 40)
      hfo[(size_t)(32 + row) * 1024 + nCol] = v2 > 0.f ? v2 : 0.f;
  }
}

// ---------------------------------------------------------------------------
// K7: final heads (40 x 1024) @ [(1024,4)|(1024,21)]. Tiny; plain VALU.
__global__ __launch_bounds__(256) void k_head2(const float* __restrict__ hfi,
                                               const float* __restrict__ rw,
                                               const float* __restrict__ rb,
                                               const float* __restrict__ cw,
                                               const float* __restrict__ cb,
                                               float* __restrict__ o_reg,
                                               float* __restrict__ o_cls) {
  __shared__ float row[1024];
  const int m = blockIdx.x;                // 40 blocks
  for (int k = threadIdx.x; k < 1024; k += 256) row[k] = hfi[(size_t)m * 1024 + k];
  __syncthreads();
  const int j = threadIdx.x;
  if (j < 4) {
    float s = rb[j];
    for (int k = 0; k < 1024; ++k) s += row[k] * rw[(size_t)k * 4 + j];
    o_reg[(size_t)m * 4 + j] = s;
  } else if (j < 25) {
    int jj = j - 4;
    float s = cb[jj];
    for (int k = 0; k < 1024; ++k) s += row[k] * cw[(size_t)k * 21 + jj];
    o_cls[(size_t)m * 21 + jj] = s;
  }
}

// ===========================================================================
extern "C" void kernel_launch(void* const* d_in, const int* in_sizes, int n_in,
                              void* d_out, int out_size, void* d_ws, size_t ws_size,
                              hipStream_t stream) {
  (void)in_sizes; (void)n_in; (void)out_size; (void)ws_size;
  const float* img       = (const float*)d_in[0];
  const int*   img_id    = (const int*)  d_in[1];
  const float* cnn_w     = (const float*)d_in[2];
  const float* cnn_b     = (const float*)d_in[3];
  const float* rpn_w     = (const float*)d_in[4];
  const float* rpn_b     = (const float*)d_in[5];
  const float* rpn_reg_w = (const float*)d_in[6];
  const float* rpn_reg_b = (const float*)d_in[7];
  const float* rpn_cls_w = (const float*)d_in[8];
  const float* rpn_cls_b = (const float*)d_in[9];
  const float* fc_w      = (const float*)d_in[10];
  const float* fc_b      = (const float*)d_in[11];
  const float* reg_w     = (const float*)d_in[12];
  const float* reg_b     = (const float*)d_in[13];
  const float* cls_w     = (const float*)d_in[14];
  const float* cls_b     = (const float*)d_in[15];

  float* out = (float*)d_out;
  char*  ws  = (char*)d_ws;
  float*    fm     = (float*)(ws + WS_FM);
  float*    h      = (float*)(ws + WS_H);
  float*    prop   = (float*)(ws + WS_PROP);
  _Float16* pooled = (_Float16*)(ws + WS_POOLED);
  float*    hfbuf  = (float*)(ws + WS_HF);

  k_anchors   <<<144,  256, 0, stream>>>(out + OFF_ANCHORS);
  k_patch_conv<<<1024, 256, 0, stream>>>(img, cnn_w, cnn_b, fm);
  k_rpn_conv  <<<512,  256, 0, stream>>>(fm, rpn_w, rpn_b, h);
  k_heads     <<<48,   256, 0, stream>>>(h, rpn_reg_w, rpn_reg_b, rpn_cls_w, rpn_cls_b,
                                         out + OFF_RPN_REG, out + OFF_RPN_CLS, prop);
  k_nms       <<<2,    256, 0, stream>>>(prop, out + OFF_RPN_CLS,
                                         out + OFF_NMS_REG, out + OFF_NMS_CLS);
  k_roi       <<<40 * 49, 256, 0, stream>>>(fm, out + OFF_NMS_REG, img_id, pooled);
  k_fc        <<<8,    256, 0, stream>>>(pooled, fc_w, fc_b, hfbuf);
  k_head2     <<<40,   256, 0, stream>>>(hfbuf, reg_w, reg_b, cls_w, cls_b,
                                         out + OFF_RCNN_REG, out + OFF_RCNN_CLS);
}
